// STGeoModule_39831526703213
// MI455X (gfx1250) — matrix operand order repaired and checked by reading, not compiled, not run on hardware
//
#include <hip/hip_runtime.h>
#include <hip/hip_bf16.h>
#include <stdint.h>

typedef _Float16 half_t;
typedef __attribute__((ext_vector_type(16))) _Float16 v16h;
typedef __attribute__((ext_vector_type(8)))  float    v8f;
typedef uint32_t u32;
typedef unsigned int u32x4 __attribute__((ext_vector_type(4)));
typedef int          i32x8 __attribute__((ext_vector_type(8)));
typedef int          i32x4 __attribute__((ext_vector_type(4)));

#ifdef __has_builtin
#  if __has_builtin(__builtin_amdgcn_tensor_load_to_lds)
#    define USE_TDM 1
#  endif
#endif
#ifndef USE_TDM
#  define USE_TDM 0
#endif

union F16Frag { v16h v; u32 u[8]; half_t h[16]; };

__device__ __forceinline__ v8f wmma_f16(v16h a, v16h b, v8f c) {
  return __builtin_amdgcn_wmma_f32_16x16x32_f16(false, a, false, b, (short)0, c, false, false);
}

// ---------------------------------------------------------------------------
// Generic conv as implicit GEMM via WMMA.  Cout fixed = 32.
// Input layout 0: [B][C][T][H][W]   layout 1: [B][T][C][H][W]
// Output: [B][32][Tout][H][W].
// Block: 256 thr = 8 waves; tile 32oc x 256pos; each wave owns 32 columns
// (2 N-subtiles) -> 4 WMMAs per K-chunk stage.
// ---------------------------------------------------------------------------
__global__ __launch_bounds__(256) void conv_wmma_k(
    const float* __restrict__ x, const float* __restrict__ w,
    const float* __restrict__ bias, float* __restrict__ out,
    int Bn, int Cin, int Tin, int Tout, int H, int W,
    int KT, int KH, int KW, int padT, int padH, int padW,
    int in_layout, int do_relu)
{
  __shared__ half_t As[32][32];     // [oc][ic]
  __shared__ half_t Bs[256][34];    // [pos][ic]  (transposed B, +2 pad)
  __shared__ u32    pc[256];

  const int tid  = threadIdx.x;
  const int lane = tid & 31;
  const int wv   = tid >> 5;
  const int r    = lane & 15;
  const int hi   = lane >> 4;
  const int HW   = H * W;
  const int Npos = Bn * Tout * HW;
  const int posBase = blockIdx.x * 256;
  const int colA = wv * 32 + r;
  const int colB = colA + 16;

  {
    int idx = posBase + tid;
    u32 valid = (idx < Npos) ? 1u : 0u;
    if (!valid) idx = 0;
    int xx = idx % W;
    int t1 = idx / W;
    int yy = t1 % H;
    int t2 = t1 / H;
    int tt = t2 % Tout;
    int bb = t2 / Tout;
    pc[tid] = (valid << 31) | ((u32)bb << 15) | ((u32)tt << 12) | ((u32)yy << 6) | (u32)xx;
  }

  v8f acc0a = {}; v8f acc1a = {};   // oc 0-15 / 16-31 x colA
  v8f acc0b = {}; v8f acc1b = {};   // oc 0-15 / 16-31 x colB
  const int chunks = Cin >> 5;

  for (int kt = 0; kt < KT; ++kt)
   for (int kh = 0; kh < KH; ++kh)
    for (int kw = 0; kw < KW; ++kw)
     for (int cc = 0; cc < chunks; ++cc) {
      __syncthreads();
      // stage A: weight tap [32oc][32ic]
      for (int i = tid; i < 1024; i += 256) {
        int oc = i >> 5, ic = i & 31;
        size_t widx = (((size_t)oc * Cin + (cc * 32 + ic)) * KT + kt) * (size_t)(KH * KW)
                      + (size_t)kh * KW + kw;
        As[oc][ic] = (half_t)w[widx];
      }
      // stage B: gathered/padded input [256pos][32ic]
      for (int i = tid; i < 8192; i += 256) {
        int ic = i >> 8;
        int p  = i & 255;
        u32 c  = pc[p];
        int xi = (int)(c & 63u) + kw - padW;
        int yi = (int)((c >> 6) & 63u) + kh - padH;
        int ti = (int)((c >> 12) & 7u) + kt - padT;
        int bb = (int)((c >> 15) & 255u);
        float vv = 0.0f;
        if ((c >> 31) && (unsigned)xi < (unsigned)W && (unsigned)yi < (unsigned)H &&
            (unsigned)ti < (unsigned)Tin) {
          int ci = cc * 32 + ic;
          size_t off = (in_layout == 0)
            ? ((((size_t)bb * Cin + ci) * Tin + ti) * HW + (size_t)yi * W + xi)
            : ((((size_t)bb * Tin + ti) * Cin + ci) * HW + (size_t)yi * W + xi);
          vv = x[off];
        }
        Bs[p][ic] = (half_t)vv;
      }
      __syncthreads();

      F16Frag af0, af1, bfA, bfB;
      const u32* a0p = (const u32*)&As[r][0];
      const u32* a1p = (const u32*)&As[16 + r][0];
      const u32* bpA = (const u32*)&Bs[colA][0];
      const u32* bpB = (const u32*)&Bs[colB][0];
#pragma unroll
      for (int j = 0; j < 8; ++j) {
        int adw = hi * 8 + j;
        af0.u[j] = a0p[adw];
        af1.u[j] = a1p[adw];
        bfA.u[j] = bpA[hi * 8 + j];
        bfB.u[j] = bpB[hi * 8 + j];
      }
      acc0a = wmma_f16(af0.v, bfA.v, acc0a);
      acc1a = wmma_f16(af1.v, bfA.v, acc1a);
      acc0b = wmma_f16(af0.v, bfB.v, acc0b);
      acc1b = wmma_f16(af1.v, bfB.v, acc1b);
     }

  // epilogue: D layout -> row = i + 8*hi, col = r (per 16x16 tile)
#pragma unroll
  for (int half = 0; half < 2; ++half) {
    int colX = half ? colB : colA;
    v8f a0 = half ? acc0b : acc0a;
    v8f a1 = half ? acc1b : acc1a;
    u32 c = pc[colX];
    if (c >> 31) {
      int xx = (int)(c & 63u), yy = (int)((c >> 6) & 63u);
      int tt = (int)((c >> 12) & 7u), bb = (int)((c >> 15) & 255u);
      size_t sp = (size_t)yy * W + xx;
#pragma unroll
      for (int i = 0; i < 8; ++i) {
        int oc = i + 8 * hi;
        float v0 = a0[i] + bias[oc];
        float v1 = a1[i] + bias[oc + 16];
        if (do_relu) { v0 = fmaxf(v0, 0.f); v1 = fmaxf(v1, 0.f); }
        out[(((size_t)bb * 32 + oc)      * Tout + tt) * HW + sp] = v0;
        out[(((size_t)bb * 32 + oc + 16) * Tout + tt) * HW + sp] = v1;
      }
    }
  }
}

// ---------------------------------------------------------------------------
// C[M,N] = relu(A[M,K] @ B[K,N]); A,B,C row-major fp32, f16 WMMA compute.
// B tile (32x256 f32) staged with the Tensor Data Mover when available.
// Block tile: 32 rows x 256 cols; grid = (N/256, M/32); 4 WMMAs/wave/stage.
// ---------------------------------------------------------------------------
__global__ __launch_bounds__(256) void gemm_relu_wmma_k(
    const float* __restrict__ A, const float* __restrict__ Bm,
    float* __restrict__ C, int M, int N, int K)
{
  __shared__ half_t As[32][34];
  __shared__ float  Bsf[32][256];
  const int tid = threadIdx.x, lane = tid & 31, wv = tid >> 5;
  const int r = lane & 15, hi = lane >> 4;
  const int m0 = blockIdx.y * 32;
  const int n0 = blockIdx.x * 256;
  const int colA = wv * 32 + r;
  const int colB = colA + 16;
  v8f acc0a = {}; v8f acc1a = {};
  v8f acc0b = {}; v8f acc1b = {};

  for (int k0 = 0; k0 < K; k0 += 32) {
    __syncthreads();
    for (int i = tid; i < 1024; i += 256) {
      int mm = i >> 5, kk = i & 31;
      As[mm][kk] = (half_t)A[(size_t)(m0 + mm) * K + (k0 + kk)];
    }
#if USE_TDM
    if (tid < 32) {  // one wave issues the tensor DMA (EXEC ignored by TDM)
      u32 ldsOff = (u32)(uintptr_t)(&Bsf[0][0]);
      uint64_t ga = (uint64_t)(uintptr_t)(Bm + (size_t)k0 * N + n0);
      // D# group0: count=1 | lds_addr | global_addr[56:0] | type=2
      u32x4 g0 = { 1u, ldsOff, (u32)ga,
                   ((u32)(ga >> 32) & 0x01FFFFFFu) | 0x80000000u };
      // D# group1: data_size=4B; tensor_dim0=256, tensor_dim1=32;
      // tile_dim0=256, tile_dim1=32; tensor_dim0_stride=N
      i32x8 g1 = { (int)(2u << 16),
                   (int)(256u << 16),
                   (int)(32u << 16),
                   (int)(256u << 16),
                   32,
                   (int)(u32)N,
                   0, 0 };
      i32x4 gz4 = { 0, 0, 0, 0 };
      i32x8 gz8 = { 0, 0, 0, 0, 0, 0, 0, 0 };
      // 6-arg form (clang-23 / therock headers): (g0, g1, g2, g3, g4, cpol)
      __builtin_amdgcn_tensor_load_to_lds(g0, g1, gz4, gz4, gz8, 0);
      __builtin_amdgcn_s_wait_tensorcnt(0);
    }
#else
    for (int i = tid; i < 8192; i += 256) {
      int kk = i >> 8, nn = i & 255;
      Bsf[kk][nn] = Bm[(size_t)(k0 + kk) * N + (n0 + nn)];
    }
#endif
    __syncthreads();

    F16Frag af0, af1, bfA, bfB;
    const u32* a0p = (const u32*)&As[r][0];
    const u32* a1p = (const u32*)&As[16 + r][0];
#pragma unroll
    for (int j = 0; j < 8; ++j) {
      int adw = hi * 8 + j;
      af0.u[j] = a0p[adw];
      af1.u[j] = a1p[adw];
    }
#pragma unroll
    for (int h2 = 0; h2 < 16; ++h2) {
      bfA.h[h2] = (half_t)Bsf[hi * 16 + h2][colA];
      bfB.h[h2] = (half_t)Bsf[hi * 16 + h2][colB];
    }
    acc0a = wmma_f16(af0.v, bfA.v, acc0a);
    acc1a = wmma_f16(af1.v, bfA.v, acc1a);
    acc0b = wmma_f16(af0.v, bfB.v, acc0b);
    acc1b = wmma_f16(af1.v, bfB.v, acc1b);
  }

#pragma unroll
  for (int half = 0; half < 2; ++half) {
    int cn = n0 + (half ? colB : colA);
    v8f a0 = half ? acc0b : acc0a;
    v8f a1 = half ? acc1b : acc1a;
#pragma unroll
    for (int i = 0; i < 8; ++i) {
      int mA = m0 + i + 8 * hi;
      C[(size_t)mA * N + cn]        = fmaxf(a0[i], 0.f);
      C[(size_t)(mA + 16) * N + cn] = fmaxf(a1[i], 0.f);
    }
  }
}

// ---------------------------------------------------------------------------
// Final FC: out[b][g][yx] = sum_d xin[b][d][yx]*fw[g][d] + fb[g]
// M = B*1024 rows (gathered NCHW), K=32, N=64.  128 rows / block.
// ---------------------------------------------------------------------------
__global__ __launch_bounds__(256) void fc_out_wmma_k(
    const float* __restrict__ xin, const float* __restrict__ fw,
    const float* __restrict__ fb, float* __restrict__ out)
{
  __shared__ half_t As[128][34];
  __shared__ half_t Bt[64][34];
  const int tid = threadIdx.x, lane = tid & 31, wv = tid >> 5;
  const int r = lane & 15, hi = lane >> 4;
  const int m0 = blockIdx.x * 128;

  for (int i = tid; i < 128 * 32; i += 256) {
    int row = i >> 5, k = i & 31;
    int g = m0 + row;
    int bb = g >> 10, yx = g & 1023;           // HW = 1024
    As[row][k] = (half_t)xin[((size_t)bb * 32 + k) * 1024 + yx];
  }
  for (int i = tid; i < 64 * 32; i += 256) {
    int n = i >> 5, k = i & 31;
    Bt[n][k] = (half_t)fw[(size_t)n * 32 + k];
  }
  __syncthreads();

  F16Frag af;
  const u32* ap = (const u32*)&As[wv * 16 + r][0];
#pragma unroll
  for (int j = 0; j < 8; ++j) {
    int adw = hi * 8 + j;
    af.u[j] = ap[adw];
  }
#pragma unroll
  for (int nt = 0; nt < 4; ++nt) {
    F16Frag bf;
    const u32* bp = (const u32*)&Bt[nt * 16 + r][0];
#pragma unroll
    for (int j = 0; j < 8; ++j) bf.u[j] = bp[hi * 8 + j];
    v8f acc = {};
    acc = wmma_f16(af.v, bf.v, acc);
#pragma unroll
    for (int i = 0; i < 8; ++i) {
      int g   = nt * 16 + r;
      int row = m0 + wv * 16 + i + 8 * hi;
      int bb = row >> 10, yx = row & 1023;
      out[((size_t)bb * 64 + g) * 1024 + yx] = acc[i] + fb[g];
    }
  }
}

// ---------------------------------------------------------------------------
// Attention helpers
// ---------------------------------------------------------------------------
__global__ __launch_bounds__(256) void satten_reduce_k(
    const float* __restrict__ k, const float* __restrict__ v,
    float* __restrict__ s, int HW)
{
  int bct = blockIdx.x;
  const float* kp = k + (size_t)bct * HW;
  const float* vp = v + (size_t)bct * HW;
  float p = 0.f;
  for (int i = threadIdx.x; i < HW; i += 256) p += kp[i] * vp[i];
#pragma unroll
  for (int m = 16; m > 0; m >>= 1) p += __shfl_xor(p, m, 32);
  __shared__ float red[8];
  if ((threadIdx.x & 31) == 0) red[threadIdx.x >> 5] = p;
  __syncthreads();
  if (threadIdx.x == 0) {
    float t = 0.f;
#pragma unroll
    for (int i = 0; i < 8; ++i) t += red[i];
    s[bct] = t;
  }
}

__global__ __launch_bounds__(256) void satten_apply_k(
    const float* __restrict__ q, const float* __restrict__ x,
    const float* __restrict__ s, const float* __restrict__ gamma,
    float* __restrict__ out, int hwShift, size_t N)
{
  size_t i = (size_t)blockIdx.x * 256 + threadIdx.x;
  if (i >= N) return;
  size_t bct = i >> hwShift;
  out[i] = gamma[0] * q[i] * s[bct] + x[i];
}

__global__ __launch_bounds__(256) void tatten_energy_k(
    const float* __restrict__ q, const float* __restrict__ k,
    float* __restrict__ e, int HW)
{
  const int tid = threadIdx.x;
  size_t base = (size_t)blockIdx.x * 7 * HW;
  float acc[49];
#pragma unroll
  for (int j = 0; j < 49; ++j) acc[j] = 0.f;
  for (int n = tid; n < HW; n += 256) {
    float qv[7], kv[7];
#pragma unroll
    for (int t = 0; t < 7; ++t) {
      qv[t] = q[base + (size_t)t * HW + n];
      kv[t] = k[base + (size_t)t * HW + n];
    }
#pragma unroll
    for (int t = 0; t < 7; ++t)
#pragma unroll
      for (int s2 = 0; s2 < 7; ++s2) acc[t * 7 + s2] += qv[t] * kv[s2];
  }
  __shared__ float ered[49];
  if (tid < 49) ered[tid] = 0.f;
  __syncthreads();
#pragma unroll
  for (int j = 0; j < 49; ++j) {
    float p = acc[j];
#pragma unroll
    for (int m = 16; m > 0; m >>= 1) p += __shfl_xor(p, m, 32);
    if ((tid & 31) == 0) atomicAdd(&ered[j], p);
  }
  __syncthreads();
  if (tid < 49) e[(size_t)blockIdx.x * 49 + tid] = ered[tid];
}

__global__ __launch_bounds__(256) void tatten_apply_k(
    const float* __restrict__ v, const float* __restrict__ x,
    const float* __restrict__ e, const float* __restrict__ gamma,
    float* __restrict__ out, int hwShift, int total)
{
  int i = blockIdx.x * 256 + threadIdx.x;
  if (i >= total) return;
  int HW = 1 << hwShift;
  int n  = i & (HW - 1);
  int bc = i >> hwShift;
  size_t base = (size_t)bc * 7 * HW;
  const float* ep = e + (size_t)bc * 49;
  float vv[7];
#pragma unroll
  for (int s2 = 0; s2 < 7; ++s2) vv[s2] = v[base + (size_t)s2 * HW + n];
  float g = gamma[0];
#pragma unroll
  for (int t = 0; t < 7; ++t) {
    float o = 0.f;
#pragma unroll
    for (int s2 = 0; s2 < 7; ++s2) o += ep[t * 7 + s2] * vv[s2];
    out[base + (size_t)t * HW + n] = g * o + x[base + (size_t)t * HW + n];
  }
}

// channel-interleave: out[b][2d][yx]=a[b][d][yx]; out[b][2d+1][yx]=bsrc[b][d][yx]
__global__ __launch_bounds__(256) void interleave_k(
    const float* __restrict__ a, const float* __restrict__ bsrc,
    float* __restrict__ o, int total, int hwShift)
{
  int i = blockIdx.x * 256 + threadIdx.x;
  if (i >= total) return;
  int HW = 1 << hwShift;
  int yx = i & (HW - 1);
  int rest = i >> hwShift;
  int d = rest & 31;
  int bb = rest >> 5;
  size_t ob = ((size_t)bb * 64 + 2 * d) * HW + yx;
  o[ob]      = a[i];
  o[ob + HW] = bsrc[i];
}

// ---------------------------------------------------------------------------
// Host-side orchestration
// ---------------------------------------------------------------------------
enum {
  IN_GRID = 0, IN_SMALL = 1, IN_TTF = 2, IN_TRANS = 3,
  P_TD2_b = 4, P_TD2_w = 5, P_TD3_b = 6, P_TD3_w = 7, P_TD_b = 8, P_TD_w = 9,
  P_WD2_b = 10, P_WD2_w = 11,
  // d1s: gamma,k_b,k_w,q_b,q_w,v_b,v_w,conv_b,conv_w
  P_D1S = 12,
  P_D2S1 = 21,
  P_D2S2 = 30,
  P_FC_b = 43, P_FC_w = 44,
  P_ST_conv_b = 45, P_ST_conv_w = 46,
  P_STS = 47,   // gamma,k_b,k_w,q_b,q_w,v_b,v_w
  P_STT = 54
};

static inline void convl(hipStream_t st, const float* x, const float* w, const float* b,
                         float* o, int Bn, int Cin, int Tin, int Tout, int H, int W,
                         int KT, int KH, int KW, int pT, int pH, int pW,
                         int lay, int relu)
{
  int Np = Bn * Tout * H * W;
  conv_wmma_k<<<(Np + 255) / 256, 256, 0, st>>>(x, w, b, o, Bn, Cin, Tin, Tout, H, W,
                                                KT, KH, KW, pT, pH, pW, lay, relu);
}

static void stream_pipeline(hipStream_t st, const float* xin, int layout0,
                            int H, int W, float* b0, float* b1, float* b2,
                            float* outK, float* sbuf, float* ebuf,
                            void* const* d_in)
{
  auto F = [&](int i) { return (const float*)d_in[i]; };
  const int Bn = 32, T = 7, D = 32;
  const int HW = H * W;
  const int shift = (HW == 1024) ? 10 : 8;

  // TD conv (1,3,3)
  convl(st, xin, F(P_TD_w), F(P_TD_b), b0, Bn, D, T, T, H, W, 1, 3, 3, 0, 1, 1, layout0, 0);
  // d3_s_atten
  convl(st, b0, F(P_STS + 2), F(P_STS + 1), b1, Bn, D, T, T, H, W, 1, 3, 3, 0, 1, 1, 0, 0); // k
  convl(st, b0, F(P_STS + 6), F(P_STS + 5), b2, Bn, D, T, T, H, W, 1, 3, 3, 0, 1, 1, 0, 0); // v
  satten_reduce_k<<<Bn * D * T, 256, 0, st>>>(b1, b2, sbuf, HW);
  convl(st, b0, F(P_STS + 4), F(P_STS + 3), b1, Bn, D, T, T, H, W, 1, 3, 3, 0, 1, 1, 0, 0); // q
  {
    size_t Ntot = (size_t)Bn * D * T * HW;
    satten_apply_k<<<(unsigned)(Ntot / 256), 256, 0, st>>>(b1, b0, sbuf, F(P_STS + 0), b2, shift, Ntot);
  }
  // conv (3,3,3) + relu
  convl(st, b2, F(P_ST_conv_w), F(P_ST_conv_b), b0, Bn, D, T, T, H, W, 3, 3, 3, 1, 1, 1, 0, 1);
  // d3_t_atten
  convl(st, b0, F(P_STT + 4), F(P_STT + 3), b1, Bn, D, T, T, H, W, 3, 1, 1, 1, 0, 0, 0, 0); // q
  convl(st, b0, F(P_STT + 2), F(P_STT + 1), b2, Bn, D, T, T, H, W, 3, 1, 1, 1, 0, 0, 0, 0); // k
  tatten_energy_k<<<Bn * D, 256, 0, st>>>(b1, b2, ebuf, HW);
  convl(st, b0, F(P_STT + 6), F(P_STT + 5), b1, Bn, D, T, T, H, W, 3, 1, 1, 1, 0, 0, 0, 0); // v
  {
    int tot = Bn * D * HW;
    tatten_apply_k<<<tot / 256, 256, 0, st>>>(b1, b0, ebuf, F(P_STT + 0), b2, shift, tot);
  }
  // TD3 conv (7,3,3), Tout = 1
  convl(st, b2, F(P_TD3_w), F(P_TD3_b), outK, Bn, D, T, 1, H, W, 7, 3, 3, 0, 1, 1, 0, 0);
}

static void d2_block(hipStream_t st, float* cur, float* t1, float* t2, float* sbuf,
                     int H, int W, int pbase, void* const* d_in)
{
  auto F = [&](int i) { return (const float*)d_in[i]; };
  const int Bn = 32, D = 32;
  const int HW = H * W;
  const int shift = (HW == 1024) ? 10 : 8;
  convl(st, cur, F(pbase + 2), F(pbase + 1), t1, Bn, D, 1, 1, H, W, 1, 3, 3, 0, 1, 1, 0, 0); // k
  convl(st, cur, F(pbase + 6), F(pbase + 5), t2, Bn, D, 1, 1, H, W, 1, 3, 3, 0, 1, 1, 0, 0); // v
  satten_reduce_k<<<Bn * D, 256, 0, st>>>(t1, t2, sbuf, HW);
  convl(st, cur, F(pbase + 4), F(pbase + 3), t1, Bn, D, 1, 1, H, W, 1, 3, 3, 0, 1, 1, 0, 0); // q
  size_t Ntot = (size_t)Bn * D * HW;
  satten_apply_k<<<(unsigned)(Ntot / 256), 256, 0, st>>>(t1, cur, sbuf, F(pbase + 0), t2, shift, Ntot);
  convl(st, t2, F(pbase + 8), F(pbase + 7), cur, Bn, D, 1, 1, H, W, 1, 3, 3, 0, 1, 1, 0, 1); // conv+relu
}

extern "C" void kernel_launch(void* const* d_in, const int* in_sizes, int n_in,
                              void* d_out, int out_size, void* d_ws, size_t ws_size,
                              hipStream_t stream)
{
  (void)in_sizes; (void)n_in; (void)out_size; (void)ws_size;
  auto F = [&](int i) { return (const float*)d_in[i]; };

  const size_t L = 7340032, S = 1835008, P = 1048576, Ps = 262144;
  float* w   = (float*)d_ws;
  float* bL0 = w;          float* bL1 = bL0 + L;  float* bL2 = bL1 + L;
  float* bS0 = bL2 + L;    float* bS1 = bS0 + S;  float* bS2 = bS1 + S;
  float* cK  = bS2 + S;        // [32,32,1024]
  float* cKs = cK + P;         // [32,32,256] -> also "ys" activation
  float* st1 = cKs + Ps;       // [32,32,1024]
  float* stk = st1 + P;        // [32,64,1024]
  float* y0  = stk + 2 * P;
  float* y1  = y0 + P;
  float* y2  = y1 + P;
  float* ys1 = y2 + P;
  float* ys2 = ys1 + Ps;
  float* sbuf = ys2 + Ps;      // 7168
  float* ebuf = sbuf + 7168;   // 50176

  // big grid stream (input layout [B][T][C][H][W])
  stream_pipeline(stream, F(IN_GRID), 1, 32, 32, bL0, bL1, bL2, cK, sbuf, ebuf, d_in);
  // small grid stream (input layout [B][C][T][H][W])
  stream_pipeline(stream, F(IN_SMALL), 0, 16, 16, bS0, bS1, bS2, cKs, sbuf, ebuf, d_in);

  // small_trans = relu(cKs[1024,256] @ trans[256,1024])
  gemm_relu_wmma_k<<<dim3(4, 32), 256, 0, stream>>>(cKs, F(IN_TRANS), st1, 1024, 1024, 256);
  // stack(conv_kk, small_trans) -> [B,64,W,H]
  interleave_k<<<(32 * 32 * 1024) / 256, 256, 0, stream>>>(cK, st1, stk, 32 * 32 * 1024, 10);
  // TD2 1x1 conv, Cin=64 -> y0
  convl(stream, stk, F(P_TD2_w), F(P_TD2_b), y0, 32, 64, 1, 1, 32, 32, 1, 1, 1, 0, 0, 0, 0, 0);

  // three d2 blocks on y (32x32) and ys (16x16, lives in cKs)
  const int order[3] = { P_D2S1, P_D2S2, P_D1S };
  for (int b = 0; b < 3; ++b) {
    d2_block(stream, y0,  y1,  y2,  sbuf, 32, 32, order[b], d_in);
    d2_block(stream, cKs, ys1, ys2, sbuf, 16, 16, order[b], d_in);
  }

  // small_trans1 = relu(ys[1024,256] @ trans)
  gemm_relu_wmma_k<<<dim3(4, 32), 256, 0, stream>>>(cKs, F(IN_TRANS), st1, 1024, 1024, 256);
  // stack(y, small_trans1) -> [B,64,W,H]
  interleave_k<<<(32 * 32 * 1024) / 256, 256, 0, stream>>>(y0, st1, stk, 32 * 32 * 1024, 10);
  // WD2 1x1 conv -> conv_kk1 in y1
  convl(stream, stk, F(P_WD2_w), F(P_WD2_b), y1, 32, 64, 1, 1, 32, 32, 1, 1, 1, 0, 0, 0, 0, 0);

  // final FC (the att/softmax path is identity): out[b][g][yx]
  fc_out_wmma_k<<<256, 256, 0, stream>>>(y1, F(P_FC_w), F(P_FC_b), (float*)d_out);
}
